// Detector_89507118449134
// MI455X (gfx1250) — matrix operand order repaired and checked by reading, not compiled, not run on hardware
//
#include <hip/hip_runtime.h>

#define Bsz   65536
#define DIN   512
#define Hdim  256
#define DLAT  128
#define Kc    1000
#define Kpad  1008
#define MADT  3.5f

// Flip to 0 if the assembler rejects the gfx1250 async-to-LDS mnemonics.
#define USE_ASYNC_LDS 1

typedef __attribute__((ext_vector_type(16))) __bf16 v16bf;
typedef __attribute__((ext_vector_type(8)))  float  v8f;

union AFrag { v16bf v; unsigned u[8]; };

__device__ inline unsigned pack2bf(float a, float b) {
  union { __bf16 h[2]; unsigned u; } p;
  p.h[0] = (__bf16)a; p.h[1] = (__bf16)b;
  return p.u;
}

// 16-byte global -> LDS copy. Async path uses the CDNA5 ASYNCcnt-tracked
// DMA (global_load_async_to_lds_b128); 'l' is a generic pointer to __shared__
// whose low 32 bits are the LDS byte offset on amdgcn.
__device__ __forceinline__ void copy16_to_lds(const void* g, void* l) {
#if USE_ASYNC_LDS
  unsigned loff = (unsigned)(unsigned long long)l;
  asm volatile("global_load_async_to_lds_b128 %0, %1, off"
               :: "v"(loff), "v"(g) : "memory");
#else
  *(uint4*)l = *(const uint4*)g;
#endif
}

__device__ __forceinline__ void async_copies_wait() {
#if USE_ASYNC_LDS
  asm volatile("s_wait_asynccnt 0" ::: "memory");
#endif
}

// ---------------------------------------------------------------------------
// Kernel 0: weight prep. W1T[H][DIN] bf16, W2T[DLAT][H] bf16,
// centB[Kpad][DLAT] bf16 (zero padded), c2[Kpad] = ||c_j||^2 (pad = +huge).
// ---------------------------------------------------------------------------
__global__ __launch_bounds__(256) void prep_kernel(
    const float* __restrict__ W1, const float* __restrict__ W2,
    const float* __restrict__ cent,
    __bf16* __restrict__ W1T, __bf16* __restrict__ W2T,
    __bf16* __restrict__ centB, float* __restrict__ c2)
{
  const int idx = blockIdx.x * blockDim.x + threadIdx.x;
  if (idx < Hdim * DIN) {                       // W1 [DIN][H] -> W1T [H][DIN]
    int h = idx / DIN, k = idx % DIN;
    W1T[idx] = (__bf16)W1[(size_t)k * Hdim + h];
  }
  if (idx < DLAT * Hdim) {                      // W2 [H][DLAT] -> W2T [DLAT][H]
    int d = idx / Hdim, h = idx % Hdim;
    W2T[idx] = (__bf16)W2[(size_t)h * DLAT + d];
  }
  if (idx < Kpad * DLAT) {                      // centroid rows, bf16, padded
    int j = idx / DLAT;
    centB[idx] = (__bf16)(j < Kc ? cent[idx] : 0.0f);
  }
  if (idx < Kpad) {                             // ||c_j||^2
    float s = 0.0f;
    if (idx < Kc) {
      for (int d = 0; d < DLAT; ++d) { float v = cent[(size_t)idx * DLAT + d]; s += v * v; }
    } else s = 3.0e38f;
    c2[idx] = s;
  }
}

// ---------------------------------------------------------------------------
// Kernel 1 (fused clean+noisy): H1[0]=relu(x@W1+b1), H1[1]=relu((x+noise)@W1+b1).
// x is loaded ONCE (halves HBM traffic vs two passes). Block: 512 thr = 16
// waves; tile 64 rows x 256 cols; wave = 16 rows x 64 cols x {clean,noisy}.
// A tiles (f32->bf16) staged via LDS; W1T K-slice staged via async-to-LDS DMA;
// each B fragment feeds TWO WMMAs.
// ---------------------------------------------------------------------------
__global__ __launch_bounds__(512) void enc1_kernel(
    const float* __restrict__ x, const float* __restrict__ noise,
    const __bf16* __restrict__ W1T, const float* __restrict__ b1,
    __bf16* __restrict__ H1)
{
  const int row0 = blockIdx.x * 64;
  const int tid  = threadIdx.x;
  const int lane = tid & 31;
  const int wave = tid >> 5;
  const int rg   = wave & 3;     // row group: rows rg*16 .. +15
  const int cg   = wave >> 2;    // col group: cols cg*64 .. +63
  const int nl   = lane & 15;
  const int hi   = lane >> 4;

  __shared__ unsigned lds0[64 * 20];            // clean A tile (16 dw + 4 pad)
  __shared__ unsigned lds1[64 * 20];            // noisy A tile
  __shared__ unsigned ldsB[256 * 20];           // W1T K-slice: 256 rows x 16 dw (+4 pad)

  __bf16* H1c = H1;
  __bf16* H1n = H1 + (size_t)Bsz * Hdim;

  const v8f vzero = {0.f,0.f,0.f,0.f,0.f,0.f,0.f,0.f};
  v8f acc0[4], acc1[4];
  #pragma unroll
  for (int s = 0; s < 4; ++s) { acc0[s] = vzero; acc1[s] = vzero; }

  for (int kt = 0; kt < DIN / 32; ++kt) {
    __syncthreads();                            // protect previous iter reads

    // ---- stage A tiles: 64x32 f32 (x and x+noise) -> bf16 pairs in LDS
    #pragma unroll
    for (int q = 0; q < 2; ++q) {
      int pi = q * 512 + tid;                   // 1024 bf16-pairs per tile
      int r  = pi >> 4;
      int c  = pi & 15;
      size_t base = (size_t)(row0 + r) * DIN + kt * 32 + c * 2;
      float2 xv = *(const float2*)(x + base);
      float2 nv = *(const float2*)(noise + base);
      if (kt + 1 < DIN / 32) __builtin_prefetch((const void*)(x + base + 32), 0, 3);
      lds0[r * 20 + c] = pack2bf(xv.x, xv.y);
      lds1[r * 20 + c] = pack2bf(xv.x + nv.x, xv.y + nv.y);
    }

    // ---- stage B slice: W1T[0..255][kt*32..+32] -> LDS, 16B DMA chunks
    #pragma unroll
    for (int q = 0; q < 2; ++q) {
      int ci = q * 512 + tid;                   // 1024 chunks of 16B
      int n  = ci >> 2;
      int cq = ci & 3;
      copy16_to_lds((const void*)(W1T + (size_t)n * DIN + kt * 32 + cq * 8),
                    (void*)&ldsB[n * 20 + cq * 4]);
    }
    async_copies_wait();
    __syncthreads();

    // ---- A fragments (ISA 16-bit 16x32 layout)
    AFrag af0, af1;
    const int ml = rg * 16 + nl;
    #pragma unroll
    for (int vi = 0; vi < 4; ++vi) {
      af0.u[vi]     = lds0[ml * 20 + hi * 4 + vi];
      af0.u[4 + vi] = lds0[ml * 20 + 8 + hi * 4 + vi];
      af1.u[vi]     = lds1[ml * 20 + hi * 4 + vi];
      af1.u[4 + vi] = lds1[ml * 20 + 8 + hi * 4 + vi];
    }

    // ---- 4 col subtiles; each B fragment feeds 2 WMMAs
    #pragma unroll
    for (int s = 0; s < 4; ++s) {
      AFrag bf_;
      int n = cg * 64 + s * 16 + nl;
      uint4 q0 = *(const uint4*)&ldsB[n * 20 + hi * 4];
      uint4 q1 = *(const uint4*)&ldsB[n * 20 + 8 + hi * 4];
      bf_.u[0]=q0.x; bf_.u[1]=q0.y; bf_.u[2]=q0.z; bf_.u[3]=q0.w;
      bf_.u[4]=q1.x; bf_.u[5]=q1.y; bf_.u[6]=q1.z; bf_.u[7]=q1.w;
      acc0[s] = __builtin_amdgcn_wmma_f32_16x16x32_bf16(false, af0.v, false, bf_.v,
                                                        (short)0, acc0[s], false, false);
      acc1[s] = __builtin_amdgcn_wmma_f32_16x16x32_bf16(false, af1.v, false, bf_.v,
                                                        (short)0, acc1[s], false, false);
    }
  }

  #pragma unroll
  for (int s = 0; s < 4; ++s) {
    int col = cg * 64 + s * 16 + nl;
    float bb = b1[col];
    #pragma unroll
    for (int r = 0; r < 8; ++r) {
      int m = rg * 16 + hi * 8 + r;
      float v0 = acc0[s][r] + bb; v0 = v0 > 0.f ? v0 : 0.f;
      float v1 = acc1[s][r] + bb; v1 = v1 > 0.f ? v1 : 0.f;
      H1c[(size_t)(row0 + m) * Hdim + col] = (__bf16)v0;
      H1n[(size_t)(row0 + m) * Hdim + col] = (__bf16)v1;
    }
  }
}

// ---------------------------------------------------------------------------
// Kernel 2: Z = H1[z] @ W2 + b2. z==0 -> Zb (bf16, feeds argmin GEMM);
// z==1 -> Zf (f32, feeds score). Wave = 16 rows x 128 cols, no LDS needed.
// ---------------------------------------------------------------------------
__global__ __launch_bounds__(256) void enc2_kernel(
    const __bf16* __restrict__ H1, const __bf16* __restrict__ W2T,
    const float* __restrict__ b2, __bf16* __restrict__ Zb, float* __restrict__ Zf)
{
  const int z    = blockIdx.z;
  const int lane = threadIdx.x & 31;
  const int wave = threadIdx.x >> 5;
  const int nl   = lane & 15;
  const int hi   = lane >> 4;
  const int row0 = blockIdx.x * 128 + wave * 16;

  const __bf16* H1z = H1 + (size_t)z * Bsz * Hdim;

  const v8f vzero = {0.f,0.f,0.f,0.f,0.f,0.f,0.f,0.f};
  v8f acc[8];
  #pragma unroll
  for (int s = 0; s < 8; ++s) acc[s] = vzero;

  const int m = row0 + nl;
  for (int kt = 0; kt < Hdim / 32; ++kt) {
    AFrag af;
    uint4 p0 = *(const uint4*)(H1z + (size_t)m * Hdim + kt * 32 + hi * 8);
    uint4 p1 = *(const uint4*)(H1z + (size_t)m * Hdim + kt * 32 + 16 + hi * 8);
    af.u[0]=p0.x; af.u[1]=p0.y; af.u[2]=p0.z; af.u[3]=p0.w;
    af.u[4]=p1.x; af.u[5]=p1.y; af.u[6]=p1.z; af.u[7]=p1.w;

    #pragma unroll
    for (int s = 0; s < 8; ++s) {
      AFrag bf_;
      int n = s * 16 + nl;
      const uint4* wp = (const uint4*)(W2T + (size_t)n * Hdim + kt * 32 + hi * 8);
      uint4 q0 = wp[0], q1 = wp[2];
      bf_.u[0]=q0.x; bf_.u[1]=q0.y; bf_.u[2]=q0.z; bf_.u[3]=q0.w;
      bf_.u[4]=q1.x; bf_.u[5]=q1.y; bf_.u[6]=q1.z; bf_.u[7]=q1.w;
      acc[s] = __builtin_amdgcn_wmma_f32_16x16x32_bf16(false, af.v, false, bf_.v,
                                                       (short)0, acc[s], false, false);
    }
  }

  #pragma unroll
  for (int s = 0; s < 8; ++s) {
    int col = s * 16 + nl;
    float bb = b2[col];
    #pragma unroll
    for (int r = 0; r < 8; ++r) {
      int mrow = row0 + hi * 8 + r;
      float vv = acc[s][r] + bb;
      if (z == 0) Zb[(size_t)mrow * DLAT + col] = (__bf16)vv;
      else        Zf[(size_t)mrow * DLAT + col] = vv;
    }
  }
}

// ---------------------------------------------------------------------------
// Kernel 3: cls[i] = argmin_j (||c_j||^2 - 2 * z1_i . c_j)  via WMMA GEMM.
// Wave = 16 rows x all 1008 centroids (63 tiles of 16), K=128 in 4 WMMAs.
// ---------------------------------------------------------------------------
__global__ __launch_bounds__(256) void argmin_kernel(
    const __bf16* __restrict__ Zb, const __bf16* __restrict__ centB,
    const float* __restrict__ c2, int* __restrict__ cls)
{
  __shared__ float c2s[Kpad];
  const int tid = threadIdx.x;
  for (int i = tid; i < Kpad; i += 256) c2s[i] = c2[i];
  __syncthreads();

  const int lane = tid & 31;
  const int wave = tid >> 5;
  const int nl   = lane & 15;
  const int hi   = lane >> 4;
  const int row0 = blockIdx.x * 128 + wave * 16;
  const int m    = row0 + nl;

  AFrag af[4];
  #pragma unroll
  for (int kk = 0; kk < 4; ++kk) {
    uint4 p0 = *(const uint4*)(Zb + (size_t)m * DLAT + kk * 32 + hi * 8);
    uint4 p1 = *(const uint4*)(Zb + (size_t)m * DLAT + kk * 32 + 16 + hi * 8);
    af[kk].u[0]=p0.x; af[kk].u[1]=p0.y; af[kk].u[2]=p0.z; af[kk].u[3]=p0.w;
    af[kk].u[4]=p1.x; af[kk].u[5]=p1.y; af[kk].u[6]=p1.z; af[kk].u[7]=p1.w;
  }

  float best[8]; int bestj[8];
  #pragma unroll
  for (int r = 0; r < 8; ++r) { best[r] = 3.4e38f; bestj[r] = 0; }

  const v8f vzero = {0.f,0.f,0.f,0.f,0.f,0.f,0.f,0.f};
  for (int jt = 0; jt < Kpad / 16; ++jt) {
    v8f acc = vzero;
    const int j = jt * 16 + nl;
    #pragma unroll
    for (int kk = 0; kk < 4; ++kk) {
      AFrag bf_;
      const uint4* cp = (const uint4*)(centB + (size_t)j * DLAT + kk * 32 + hi * 8);
      uint4 q0 = cp[0], q1 = cp[2];
      bf_.u[0]=q0.x; bf_.u[1]=q0.y; bf_.u[2]=q0.z; bf_.u[3]=q0.w;
      bf_.u[4]=q1.x; bf_.u[5]=q1.y; bf_.u[6]=q1.z; bf_.u[7]=q1.w;
      acc = __builtin_amdgcn_wmma_f32_16x16x32_bf16(false, af[kk].v, false, bf_.v,
                                                    (short)0, acc, false, false);
    }
    const float cj = c2s[j];
    const bool ok = (j < Kc);
    #pragma unroll
    for (int r = 0; r < 8; ++r) {
      float val = cj - 2.0f * acc[r];
      if (ok && val < best[r]) { best[r] = val; bestj[r] = j; }
    }
  }

  // argmin across the 16 N-lanes (rows 0-7 live in lanes 0-15, rows 8-15 in 16-31)
  #pragma unroll
  for (int off = 1; off < 16; off <<= 1) {
    #pragma unroll
    for (int r = 0; r < 8; ++r) {
      float ov = __shfl_xor(best[r],  off, 32);
      int   oj = __shfl_xor(bestj[r], off, 32);
      if (ov < best[r] || (ov == best[r] && oj < bestj[r])) { best[r] = ov; bestj[r] = oj; }
    }
  }
  if (nl == 0) {
    #pragma unroll
    for (int r = 0; r < 8; ++r) cls[row0 + hi * 8 + r] = bestj[r];
  }
}

// ---------------------------------------------------------------------------
// Kernel 4: drift[i] = | ||z2_i - c[cls_i]|| - med[cls_i] | / mad[cls_i] > 3.5
// One wave per row; float4 loads + shfl reduction.
// ---------------------------------------------------------------------------
__global__ __launch_bounds__(256) void score_kernel(
    const float* __restrict__ Zf, const float* __restrict__ cent,
    const int* __restrict__ cls, const float* __restrict__ dmed,
    const float* __restrict__ mad, int* __restrict__ drift)
{
  const int lane = threadIdx.x & 31;
  const int wave = threadIdx.x >> 5;
  const int row  = blockIdx.x * 8 + wave;
  const int c    = cls[row];

  float4 zv = *(const float4*)(Zf   + (size_t)row * DLAT + lane * 4);
  float4 cv = *(const float4*)(cent + (size_t)c   * DLAT + lane * 4);
  float dx = zv.x - cv.x, dy = zv.y - cv.y, dz = zv.z - cv.z, dw = zv.w - cv.w;
  float s = dx*dx + dy*dy + dz*dz + dw*dw;
  #pragma unroll
  for (int off = 16; off > 0; off >>= 1) s += __shfl_xor(s, off, 32);
  if (lane == 0) {
    float dis = sqrtf(s);
    float sc  = fabsf(dis - dmed[c]) / mad[c];
    drift[row] = (sc > MADT) ? 1 : 0;
  }
}

// ---------------------------------------------------------------------------
extern "C" void kernel_launch(void* const* d_in, const int* in_sizes, int n_in,
                              void* d_out, int out_size, void* d_ws, size_t ws_size,
                              hipStream_t stream)
{
  (void)in_sizes; (void)n_in; (void)out_size; (void)ws_size;
  const float* x     = (const float*)d_in[0];
  const float* noise = (const float*)d_in[1];
  const float* W1    = (const float*)d_in[2];
  const float* b1    = (const float*)d_in[3];
  const float* W2    = (const float*)d_in[4];
  const float* b2    = (const float*)d_in[5];
  const float* cent  = (const float*)d_in[6];
  const float* dmed  = (const float*)d_in[7];
  const float* mad   = (const float*)d_in[8];
  int* drift = (int*)d_out;

  char* ws = (char*)d_ws;
  size_t off = 0;
  auto take = [&](size_t bytes) { char* p = ws + off; off = (off + bytes + 255) & ~(size_t)255; return p; };
  __bf16* W1T   = (__bf16*)take((size_t)Hdim * DIN * 2);
  __bf16* W2T   = (__bf16*)take((size_t)DLAT * Hdim * 2);
  __bf16* centB = (__bf16*)take((size_t)Kpad * DLAT * 2);
  float*  c2    = (float*) take((size_t)Kpad * 4);
  __bf16* H1    = (__bf16*)take((size_t)2 * Bsz * Hdim * 2);   // [z][B][H]
  __bf16* Zb    = (__bf16*)take((size_t)Bsz * DLAT * 2);       // clean z1 (bf16)
  float*  Zf    = (float*) take((size_t)Bsz * DLAT * 4);       // noisy z2 (f32)
  int*    cls   = (int*)   take((size_t)Bsz * 4);

  prep_kernel  <<<512, 256, 0, stream>>>(W1, W2, cent, W1T, W2T, centB, c2);
  enc1_kernel  <<<Bsz / 64, 512, 0, stream>>>(x, noise, W1T, b1, H1);
  enc2_kernel  <<<dim3(Bsz / 128, 1, 2), 256, 0, stream>>>(H1, W2T, b2, Zb, Zf);
  argmin_kernel<<<Bsz / 128, 256, 0, stream>>>(Zb, centB, c2, cls);
  score_kernel <<<Bsz / 8,   256, 0, stream>>>(Zf, cent, cls, dmed, mad, drift);
}
